// AlphaQubitLikeModel_21320217658062
// MI455X (gfx1250) — matrix operand order repaired and checked by reading, not compiled, not run on hardware
//
#include <hip/hip_runtime.h>
#include <math.h>

typedef _Float16 h16v __attribute__((ext_vector_type(16)));
typedef _Float16 h8v  __attribute__((ext_vector_type(8)));
typedef _Float16 h4v  __attribute__((ext_vector_type(4)));
typedef float    f8v  __attribute__((ext_vector_type(8)));
typedef float    f4v  __attribute__((ext_vector_type(4)));

#define WMMA(a, b, c) __builtin_amdgcn_wmma_f32_16x16x32_f16(false, (a), false, (b), (short)0, (c), false, false)

__device__ __forceinline__ float geluf(float x) {
  return 0.5f * x * (1.0f + erff(x * 0.70710678118654752f));  // exact erf (approximate=False)
}

__device__ __forceinline__ h16v joinh(h8v lo, h8v hi) {
  union { h16v v; h8v p[2]; } u; u.p[0] = lo; u.p[1] = hi; return u.v;
}

// A fragment (16x32, 16-bit) from row-major f16 LDS (contiguous K order, matches B):
// lane<16: halves 0..15 -> K 0..15 (row M=lane)
// lane>=16: halves 0..15 -> K 16..31 (row M=lane-16)
__device__ __forceinline__ h16v ldsA(const _Float16* A, int stride, int mrow, int kcol, int lane) {
  int row = mrow + (lane & 15);
  const _Float16* p = A + row * stride + kcol + ((lane < 16) ? 0 : 16);
  return joinh(*(const h8v*)p, *(const h8v*)(p + 8));
}

__device__ __forceinline__ h16v ldsA_shift(const _Float16* A, int stride, int mrow, int kcol, int shift, int lane) {
  int row = mrow + (lane & 15) + shift;
  if (row < 0 || row >= 120) { h16v z = {}; return z; }
  const _Float16* p = A + row * stride + kcol + ((lane < 16) ? 0 : 16);
  return joinh(*(const h8v*)p, *(const h8v*)(p + 8));
}

// B fragment (32x16, 16-bit) from row-major f16 LDS where LDS row index = N,
// consecutive halves = K: lane n=lane&15, halves h -> K=(lane<16?0:16)+h
__device__ __forceinline__ h16v ldsB(const _Float16* B, int stride, int nbase, int kcol, int lane) {
  int row = nbase + (lane & 15);
  const _Float16* p = B + row * stride + kcol + ((lane < 16) ? 0 : 16);
  return joinh(*(const h8v*)p, *(const h8v*)(p + 8));
}

// B fragment from packed global weights: [nt][kt][lane][16 halves]
__device__ __forceinline__ h16v gblB(const _Float16* pk, int ktiles, int nt, int kt, int lane) {
  const _Float16* p = pk + ((((size_t)nt * ktiles + kt) * 32 + lane) << 4);
  return joinh(*(const h8v*)p, *(const h8v*)(p + 8));
}

// ---------------- GEMM helpers (tiles distributed over 8 waves) ----------------
// force-inlined so literal mtiles/ntiles/ktiles constant-propagate (no int division,
// fully unrolled k-loops -> pipelined ds_load_b128 / v_wmma chains)

// D16[m][n] = act(acc + bias[n]) ; act: 0 none, 1 gelu
__device__ __forceinline__ void gemm16(const _Float16* A, int astride, const _Float16* Bpk,
                                       int mtiles, int ntiles, int ktiles, const float* bias,
                                       _Float16* D, int dstride, int act, int wid, int lane) {
  for (int tile = wid; tile < mtiles * ntiles; tile += 8) {
    int mt = tile / ntiles, nt = tile % ntiles;
    f8v acc = {};
#pragma unroll
    for (int kt = 0; kt < ktiles; ++kt) {
      h16v a = ldsA(A, astride, mt * 16, kt * 32, lane);
      h16v b = gblB(Bpk, ktiles, nt, kt, lane);
      acc = WMMA(a, b, acc);
    }
    int n = nt * 16 + (lane & 15);
    int mb = mt * 16 + ((lane < 16) ? 0 : 8);
    float bn = bias[n];
#pragma unroll
    for (int r = 0; r < 8; ++r) {
      float v = acc[r] + bn;
      if (act) v = geluf(v);
      D[(mb + r) * dstride + n] = (_Float16)v;
    }
  }
}

// transposed f16 store (used to build V^T): D[n][m]
__device__ __forceinline__ void gemm16T(const _Float16* A, int astride, const _Float16* Bpk,
                                        int mtiles, int ntiles, int ktiles, const float* bias,
                                        _Float16* D, int wid, int lane) {
  for (int tile = wid; tile < mtiles * ntiles; tile += 8) {
    int mt = tile / ntiles, nt = tile % ntiles;
    f8v acc = {};
#pragma unroll
    for (int kt = 0; kt < ktiles; ++kt) {
      h16v a = ldsA(A, astride, mt * 16, kt * 32, lane);
      h16v b = gblB(Bpk, ktiles, nt, kt, lane);
      acc = WMMA(a, b, acc);
    }
    int n = nt * 16 + (lane & 15);
    int mb = mt * 16 + ((lane < 16) ? 0 : 8);
    float bn = bias[n];
#pragma unroll
    for (int r = 0; r < 8; ++r)
      D[n * 128 + mb + r] = (_Float16)(acc[r] + bn);
  }
}

// f32 output: D[m][n] (+)= act(acc + bias[n])
__device__ __forceinline__ void gemm32(const _Float16* A, int astride, const _Float16* Bpk,
                                       int mtiles, int ntiles, int ktiles, const float* bias,
                                       float* D, int dstride, int act, bool accum, int wid, int lane) {
  for (int tile = wid; tile < mtiles * ntiles; tile += 8) {
    int mt = tile / ntiles, nt = tile % ntiles;
    f8v acc = {};
#pragma unroll
    for (int kt = 0; kt < ktiles; ++kt) {
      h16v a = ldsA(A, astride, mt * 16, kt * 32, lane);
      h16v b = gblB(Bpk, ktiles, nt, kt, lane);
      acc = WMMA(a, b, acc);
    }
    int n = nt * 16 + (lane & 15);
    int mb = mt * 16 + ((lane < 16) ? 0 : 8);
    float bn = bias[n];
#pragma unroll
    for (int r = 0; r < 8; ++r) {
      float v = acc[r] + bn;
      if (act) v = geluf(v);
      if (accum) D[(mb + r) * dstride + n] += v;
      else       D[(mb + r) * dstride + n] = v;
    }
  }
}

__device__ __forceinline__ void layernorm128(const float* src, _Float16* dst,
                                             const float* g, const float* bt,
                                             int wid, int lane) {
  for (int row = wid; row < 128; row += 8) {
    const float* x = src + row * 128;
    float s = 0.f;
    for (int c = lane; c < 128; c += 32) s += x[c];
#pragma unroll
    for (int o = 16; o > 0; o >>= 1) s += __shfl_xor(s, o, 32);
    float mean = s * (1.0f / 128.0f);
    float v = 0.f;
    for (int c = lane; c < 128; c += 32) { float d = x[c] - mean; v += d * d; }
#pragma unroll
    for (int o = 16; o > 0; o >>= 1) v += __shfl_xor(v, o, 32);
    float rstd = rsqrtf(v * (1.0f / 128.0f) + 1e-5f);
    for (int c = lane; c < 128; c += 32)
      dst[row * 128 + c] = (_Float16)((x[c] - mean) * rstd * g[c] + bt[c]);
  }
}

// ---------------- prep kernels ----------------

// Bp[l][b][h][i][j] = sum_d bias[b][i][j][d] * Wb[l][d][h]
__global__ __launch_bounds__(256) void bp_kernel(const float* __restrict__ bias,
                                                 const float* __restrict__ Wb,
                                                 float* __restrict__ Bp) {
  int idx = blockIdx.x * blockDim.x + threadIdx.x;
  if (idx >= 16 * 120 * 120) return;
  int j = idx % 120, i = (idx / 120) % 120, b = idx / 14400;
  const float* v = bias + (size_t)idx * 32;
  float vv[32];
#pragma unroll
  for (int d = 0; d < 32; ++d) vv[d] = v[d];
  for (int l = 0; l < 2; ++l)
    for (int h = 0; h < 4; ++h) {
      float s = 0.f;
#pragma unroll
      for (int d = 0; d < 32; ++d) s += vv[d] * Wb[(l * 32 + d) * 4 + h];
      Bp[((size_t)((l * 16 + b) * 4 + h)) * 14400 + i * 120 + j] = s;
    }
}

// Pack f32 source into f16 WMMA-B fragment order: idx = ((nt*ktiles+kt)*32+lane)*16+half
// src addr = (k%KI)*ski + (k/KI)*sko + (n%NI)*sni + (n/NI)*sno ; zero-pad k>=Ksrc or n>=Nsrc
__global__ __launch_bounds__(256) void pack_kernel(const float* __restrict__ src,
                                                   _Float16* __restrict__ dst,
                                                   int K, int N, int Ksrc, int Nsrc,
                                                   int KI, int ski, int sko,
                                                   int NI, int sni, int sno) {
  int ktiles = K >> 5;
  int total = K * N;
  int idx = blockIdx.x * blockDim.x + threadIdx.x;
  if (idx >= total) return;
  int half = idx & 15;
  int lane = (idx >> 4) & 31;
  int tidx = idx >> 9;
  int kt = tidx % ktiles;
  int nt = tidx / ktiles;
  int k = kt * 32 + ((lane < 16) ? 0 : 16) + half;
  int n = nt * 16 + (lane & 15);
  float v = 0.f;
  if (k < Ksrc && n < Nsrc)
    v = src[(size_t)(k % KI) * ski + (size_t)(k / KI) * sko + (size_t)(n % NI) * sni + (size_t)(n / NI) * sno];
  dst[idx] = (_Float16)v;
}

// ---------------- fused persistent forward: one workgroup per batch element ----------------

struct Params {
  const float *meas, *evnt, *leak, *evlk;
  const float *pm_w, *pm_b, *pe_w, *pe_b, *pl_w, *pl_b, *pel_w, *pel_b;
  const float *stab_emb, *cyc_emb;
  const float *er_ln_s, *er_ln_b, *er_fc1_b, *er_fc2_b;
  const float *ln1_s, *ln1_b, *bq, *bk, *bv, *bo, *ln2_s, *ln2_b;
  const float *f1_b, *f2_b, *conv_b, *lnf_s, *lnf_b, *Pvec;
  const float *sc_b, *dr_b, *rb1_b, *rb2_b, *out_w, *out_b;
  const int *stab_ids, *cycle_ids;
  const float *Bp;
  const _Float16 *er1pk, *er2pk, *wqpk, *wkpk, *wvpk, *wopk;
  const _Float16 *f1pk, *f2pk, *convpk, *scpk, *drpk, *rb1pk, *rb2pk;
  float *out;
};

__global__ __launch_bounds__(256) void fused_forward(Params p) {
  extern __shared__ __align__(16) char smem[];
  float*    X    = (float*)(smem);                 // 128x128 f32 residual (S padded 120->128)
  float*    SCR  = (float*)(smem + 65536);         // 128x128 f32 scratch (embed E / scores / dr out)
  _Float16* A16  = (_Float16*)(smem + 131072);     // 128x128 f16 LN'd activations (GEMM A)
  _Float16* Q16  = (_Float16*)(smem + 163840);     // 128x128 f16
  _Float16* K16  = (_Float16*)(smem + 196608);     // 128x128 f16 (rows=keys -> free K^T fragments)
  _Float16* V16T = (_Float16*)(smem + 229376);     // 128x128 f16 transposed V (rows=channels)
  _Float16* B16  = (_Float16*)(smem + 262144);     // 128x256 f16 (probs / Y / FFN act / readout A)
  _Float16* Y16  = B16 + 16384;
  f4v* X4   = (f4v*)X;
  f4v* SCR4 = (f4v*)SCR;

  const int b = blockIdx.x;
  const int tid = threadIdx.x;
  const int lane = tid & 31;
  const int wid = tid >> 5;

  for (int i = tid; i < 4096; i += 256) { f4v z = {}; X4[i] = z; }
  __syncthreads();

  for (int t = 0; t < 8; ++t) {
    // ---- embed(t): elementwise -> SCR (rows >=120 zero) ----
    int cid = p.cycle_ids[t];
    const float* ce = p.cyc_emb + cid * 128;
    for (int idx = tid; idx < 16384; idx += 256) {
      int s = idx >> 7, c = idx & 127;
      float e = 0.f;
      if (s < 120) {
        int base = (b * 8 + t) * 120 + s;
        float m = p.meas[base], ev = p.evnt[base], lk = p.leak[base], el = p.evlk[base];
        int sid = p.stab_ids[s];
        e = m * p.pm_w[c] + p.pm_b[c] + ev * p.pe_w[c] + p.pe_b[c]
          + lk * p.pl_w[c] + p.pl_b[c] + el * p.pel_w[c] + p.pel_b[c]
          + p.stab_emb[sid * 128 + c] + ce[c];
      }
      SCR[idx] = e;
    }
    __syncthreads();
    // ---- 2 embed residual MLP blocks ----
    for (int r = 0; r < 2; ++r) {
      layernorm128(SCR, A16, p.er_ln_s + r * 128, p.er_ln_b + r * 128, wid, lane);
      __syncthreads();
      gemm16(A16, 128, p.er1pk + r * 16384, 8, 8, 4, p.er_fc1_b + r * 128, B16, 128, 1, wid, lane);
      __syncthreads();
      gemm32(B16, 128, p.er2pk + r * 16384, 8, 8, 4, p.er_fc2_b + r * 128, SCR, 128, 0, true, wid, lane);
      __syncthreads();
    }
    // ---- X = (X + embed) / sqrt(2) ----
    for (int i = tid; i < 4096; i += 256) {
      f4v x = X4[i], e = SCR4[i];
      X4[i] = (x + e) * 0.70710678118654752f;
    }
    __syncthreads();

    for (int l = 0; l < 2; ++l) {
      // warm the WGP$/L2 with this layer's FFN + conv packed weights (global_prefetch_b8)
      {
        const char* pf1 = (const char*)(p.f1pk + l * 65536);    // 128 KB
        const char* pf2 = (const char*)(p.f2pk + l * 32768);    // 64 KB
        const char* pfc = (const char*)(p.convpk + l * 3 * 16384); // 96 KB
        for (int i = tid; i < 1024; i += 256) __builtin_prefetch(pf1 + i * 128, 0, 1);
        for (int i = tid; i < 512; i += 256)  __builtin_prefetch(pf2 + i * 128, 0, 1);
        for (int i = tid; i < 768; i += 256)  __builtin_prefetch(pfc + i * 128, 0, 1);
      }
      // ---- attention ----
      layernorm128(X, A16, p.ln1_s + l * 128, p.ln1_b + l * 128, wid, lane);
      __syncthreads();
      gemm16 (A16, 128, p.wqpk + l * 16384, 8, 8, 4, p.bq + l * 128, Q16, 128, 0, wid, lane);
      gemm16 (A16, 128, p.wkpk + l * 16384, 8, 8, 4, p.bk + l * 128, K16, 128, 0, wid, lane);
      gemm16T(A16, 128, p.wvpk + l * 16384, 8, 8, 4, p.bv + l * 128, V16T, wid, lane);
      __syncthreads();

      for (int hh = 0; hh < 4; ++hh) {
        const float* Bp_bh = p.Bp + ((size_t)((l * 16 + b) * 4 + hh)) * 14400;
        // scores: single k-step (da=32) per 16x16 tile
        for (int tile = wid; tile < 64; tile += 8) {
          int qt = tile >> 3, kt2 = tile & 7;
          h16v a = ldsA(Q16, 128, qt * 16, hh * 32, lane);
          h16v bb = ldsB(K16, 128, kt2 * 16, hh * 32, lane);
          f8v acc = {};
          acc = WMMA(a, bb, acc);
          int k = kt2 * 16 + (lane & 15);
          int qb = qt * 16 + ((lane < 16) ? 0 : 8);
#pragma unroll
          for (int r = 0; r < 8; ++r) {
            int q = qb + r;
            float bp = (q < 120 && k < 120) ? Bp_bh[q * 120 + k] : 0.f;
            float v = (acc[r] + bp) * 0.17677669529663687f;  // scale AFTER bias add
            if (k >= 120) v = -1e30f;                        // mask padded keys
            SCR[q * 128 + k] = v;
          }
        }
        __syncthreads();
        // softmax rows -> probs f16 in B16 (native v_exp_f32 path)
        for (int row = wid; row < 128; row += 8) {
          _Float16* prow = B16 + row * 128;
          if (row >= 120) {
            for (int c = lane; c < 128; c += 32) prow[c] = (_Float16)0.f;
          } else {
            const float* srow = SCR + row * 128;
            float mx = -3.0e38f;
            for (int c = lane; c < 128; c += 32) mx = fmaxf(mx, srow[c]);
#pragma unroll
            for (int o = 16; o > 0; o >>= 1) mx = fmaxf(mx, __shfl_xor(mx, o, 32));
            float vals[4]; float sum = 0.f;
#pragma unroll
            for (int i = 0; i < 4; ++i) {
              int c = lane + 32 * i;
              float e = __expf(srow[c] - mx);
              vals[i] = e; sum += e;
            }
#pragma unroll
            for (int o = 16; o > 0; o >>= 1) sum += __shfl_xor(sum, o, 32);
            float inv = 1.f / sum;
#pragma unroll
            for (int i = 0; i < 4; ++i) {
              int c = lane + 32 * i;
              prow[c] = (_Float16)((c < 120) ? vals[i] * inv : 0.f);
            }
          }
        }
        __syncthreads();
        // O = P @ V_head -> Y16 columns [hh*32, hh*32+32)
        for (int tile = wid; tile < 16; tile += 8) {
          int mt = tile >> 1, nt = tile & 1;
          f8v acc = {};
#pragma unroll
          for (int kt = 0; kt < 4; ++kt) {
            h16v a = ldsA(B16, 128, mt * 16, kt * 32, lane);
            h16v bb = ldsB(V16T, 128, hh * 32 + nt * 16, kt * 32, lane);
            acc = WMMA(a, bb, acc);
          }
          int n = hh * 32 + nt * 16 + (lane & 15);
          int mb = mt * 16 + ((lane < 16) ? 0 : 8);
#pragma unroll
          for (int r = 0; r < 8; ++r) Y16[(mb + r) * 128 + n] = (_Float16)acc[r];
        }
        __syncthreads();
      }
      // out projection: X += Y @ Wo + bo
      gemm32(Y16, 128, p.wopk + l * 16384, 8, 8, 4, p.bo + l * 128, X, 128, 0, true, wid, lane);
      __syncthreads();

      // ---- gated FFN ----
      layernorm128(X, A16, p.ln2_s + l * 128, p.ln2_b + l * 128, wid, lane);
      __syncthreads();
      const float* f1b = p.f1_b + l * 512;
      for (int tile = wid; tile < 128; tile += 8) {   // 8 mtiles x 16 ct
        int mt = tile >> 4, ct = tile & 15;
        f8v aacc = {}, gacc = {};
#pragma unroll
        for (int kt = 0; kt < 4; ++kt) {
          h16v a = ldsA(A16, 128, mt * 16, kt * 32, lane);
          h16v b1 = gblB(p.f1pk + l * 65536, 4, ct, kt, lane);
          h16v b2 = gblB(p.f1pk + l * 65536, 4, ct + 16, kt, lane);
          aacc = WMMA(a, b1, aacc);
          gacc = WMMA(a, b2, gacc);
        }
        int n = ct * 16 + (lane & 15);
        int mb = mt * 16 + ((lane < 16) ? 0 : 8);
        float ba = f1b[n], bg = f1b[256 + n];
#pragma unroll
        for (int r = 0; r < 8; ++r)
          B16[(mb + r) * 256 + n] = (_Float16)(geluf(aacc[r] + ba) * (gacc[r] + bg));
      }
      __syncthreads();
      gemm32(B16, 256, p.f2pk + l * 32768, 8, 8, 8, p.f2_b + l * 128, X, 128, 0, true, wid, lane);
      __syncthreads();

      // ---- conv block: X += gelu(conv1d_k3_same(X) + b), as 3 shifted GEMMs ----
      for (int i = tid; i < 4096; i += 256) {
        f4v v = X4[i];
        h4v hv = { (_Float16)v[0], (_Float16)v[1], (_Float16)v[2], (_Float16)v[3] };
        *(h4v*)(A16 + i * 4) = hv;
      }
      __syncthreads();
      const float* cb = p.conv_b + l * 128;
      for (int tile = wid; tile < 64; tile += 8) {
        int mt = tile >> 3, nt = tile & 7;
        f8v acc = {};
#pragma unroll
        for (int dk = 0; dk < 3; ++dk) {
#pragma unroll
          for (int kt = 0; kt < 4; ++kt) {
            h16v a = ldsA_shift(A16, 128, mt * 16, kt * 32, dk - 1, lane);
            h16v bb = gblB(p.convpk + (l * 3 + dk) * 16384, 4, nt, kt, lane);
            acc = WMMA(a, bb, acc);
          }
        }
        int n = nt * 16 + (lane & 15);
        int mb = mt * 16 + ((lane < 16) ? 0 : 8);
        float bn = cb[n];
#pragma unroll
        for (int r = 0; r < 8; ++r) {
          int q = mb + r;
          if (q < 120) X[q * 128 + n] += geluf(acc[r] + bn);
        }
      }
      __syncthreads();
    } // l
  } // t

  // ---------------- readout ----------------
  layernorm128(X, A16, p.lnf_s, p.lnf_b, wid, lane);
  __syncthreads();
  // build 36x512 A for the 2x2/stride2 grid conv (pad rows to 48)
  for (int idx = tid; idx < 48 * 512; idx += 256) {
    int pos = idx >> 9, k = idx & 511;
    float v = 0.f;
    if (pos < 36) {
      int pq = k >> 7, c = k & 127;
      int i2 = pos / 6, j2 = pos % 6, di = pq >> 1, dj = pq & 1;
      int cell = (2 * i2 + di) * 12 + (2 * j2 + dj);
      v = (cell < 120) ? (float)A16[cell * 128 + c] : p.Pvec[c];
    }
    B16[idx] = (_Float16)v;
  }
  __syncthreads();
  gemm16(B16, 512, p.scpk, 3, 8, 16, p.sc_b, A16, 128, 1, wid, lane);      // gelu(conv)
  __syncthreads();
  gemm32(A16, 128, p.drpk, 3, 3, 4, p.dr_b, SCR, 48, 1, false, wid, lane); // gelu(dim-reduce)
  __syncthreads();
  // mean over grid rows -> Xr[6][48]
  float* Xr = X;
  for (int idx = tid; idx < 288; idx += 256) {
    int j2 = idx / 48, r = idx % 48;
    float s = 0.f;
    for (int i2 = 0; i2 < 6; ++i2) s += SCR[((i2 * 6 + j2) * 48) + r];
    Xr[idx] = s * (1.0f / 6.0f);
  }
  for (int i = tid; i < 2048; i += 256) A16[i] = (_Float16)0.f;  // zero rbA + T16 pads
  __syncthreads();
  _Float16* rbA = A16;          // 16x64 (K padded 48->64)
  _Float16* T16 = A16 + 1024;   // 16x64
  for (int blk = 0; blk < 16; ++blk) {
    for (int idx = tid; idx < 288; idx += 256)
      rbA[(idx / 48) * 64 + (idx % 48)] = (_Float16)Xr[idx];
    __syncthreads();
    gemm16(rbA, 64, p.rb1pk + blk * 3072, 1, 3, 2, p.rb1_b + blk * 48, T16, 64, 1, wid, lane);
    __syncthreads();
    for (int tile = wid; tile < 3; tile += 8) {
      int nt = tile;
      f8v acc = {};
#pragma unroll
      for (int kt = 0; kt < 2; ++kt) {
        h16v a = ldsA(T16, 64, 0, kt * 32, lane);
        h16v bb = gblB(p.rb2pk + blk * 3072, 2, nt, kt, lane);
        acc = WMMA(a, bb, acc);
      }
      int n = nt * 16 + (lane & 15);
      int mb = (lane < 16) ? 0 : 8;
      float bn = p.rb2_b[blk * 48 + n];
#pragma unroll
      for (int r = 0; r < 8; ++r) {
        int m = mb + r;
        if (m < 6) Xr[m * 48 + n] += acc[r] + bn;
      }
    }
    __syncthreads();
  }
  if (tid == 0) {
    float ob = p.out_b[0];
    float accv = 0.f;
    for (int j2 = 0; j2 < 6; ++j2) {
      float s = ob;
      for (int r = 0; r < 48; ++r) s += Xr[j2 * 48 + r] * p.out_w[r];
      accv += s;
    }
    p.out[b] = accv * (1.0f / 6.0f);
  }
}

// ---------------- host ----------------

extern "C" void kernel_launch(void* const* d_in, const int* in_sizes, int n_in,
                              void* d_out, int out_size, void* d_ws, size_t ws_size,
                              hipStream_t stream) {
  (void)in_sizes; (void)n_in; (void)out_size; (void)ws_size;
  auto F = [&](int i) { return (const float*)d_in[i]; };
  auto I = [&](int i) { return (const int*)d_in[i]; };

  char* wsb = (char*)d_ws;
  float* Bp = (float*)wsb;                                   // 2*16*4*120*120 f32 = 7,372,800 B
  _Float16* pk = (_Float16*)(wsb + 7372800);
  _Float16* er1pk  = pk;               // 2 x 128x128
  _Float16* er2pk  = er1pk + 32768;
  _Float16* wqpk   = er2pk + 32768;    // 2 x 128x128 (heads concat on N)
  _Float16* wkpk   = wqpk + 32768;
  _Float16* wvpk   = wkpk + 32768;
  _Float16* wopk   = wvpk + 32768;
  _Float16* f1pk   = wopk + 32768;     // 2 x 128x512
  _Float16* f2pk   = f1pk + 131072;    // 2 x 256x128
  _Float16* convpk = f2pk + 65536;     // 2 x 3 x 128x128
  _Float16* scpk   = convpk + 98304;   // 512x128
  _Float16* drpk   = scpk + 65536;     // 128x48
  _Float16* rb1pk  = drpk + 6144;      // 16 x 64x48
  _Float16* rb2pk  = rb1pk + 49152;    // 16 x 64x48

  bp_kernel<<<(16 * 120 * 120 + 255) / 256, 256, 0, stream>>>(F(4), F(23), Bp);

  auto pack = [&](const float* src, _Float16* dst, int K, int N, int Ksrc, int Nsrc,
                  int KI, int ski, int sko, int NI, int sni, int sno) {
    int total = K * N;
    pack_kernel<<<(total + 255) / 256, 256, 0, stream>>>(src, dst, K, N, Ksrc, Nsrc,
                                                         KI, ski, sko, NI, sni, sno);
  };

  for (int r = 0; r < 2; ++r) {
    pack(F(17) + r * 16384, er1pk + r * 16384, 128, 128, 128, 128, 128, 128, 0, 128, 1, 0);
    pack(F(19) + r * 16384, er2pk + r * 16384, 128, 128, 128, 128, 128, 128, 0, 128, 1, 0);
  }
  for (int l = 0; l < 2; ++l) {
    // Wq[l][h][d][e]: n = h*32+e -> addr = k*32 + (n%32) + (n/32)*4096
    pack(F(24) + l * 16384, wqpk + l * 16384, 128, 128, 128, 128, 128, 32, 0, 32, 1, 4096);
    pack(F(26) + l * 16384, wkpk + l * 16384, 128, 128, 128, 128, 128, 32, 0, 32, 1, 4096);
    pack(F(28) + l * 16384, wvpk + l * 16384, 128, 128, 128, 128, 128, 32, 0, 32, 1, 4096);
    pack(F(30) + l * 16384, wopk + l * 16384, 128, 128, 128, 128, 128, 128, 0, 128, 1, 0);
    pack(F(34) + l * 65536, f1pk + l * 65536, 128, 512, 128, 512, 128, 512, 0, 512, 1, 0);
    pack(F(36) + l * 32768, f2pk + l * 32768, 256, 128, 256, 128, 256, 128, 0, 128, 1, 0);
    for (int dk = 0; dk < 3; ++dk)   // conv_w OIH: B[i][o] = w[o][i][dk]
      pack(F(38) + l * 49152 + dk, convpk + (l * 3 + dk) * 16384,
           128, 128, 128, 128, 128, 3, 0, 128, 384, 0);
  }
  // sc_w OIHW: k = (di*2+dj)*128 + c -> addr = (k%128)*4 + (k/128) + o*512
  pack(F(43), scpk, 512, 128, 512, 128, 128, 4, 1, 128, 512, 0);
  // dr_w (rd,d): B[c][r] = dr_w[r*128+c]
  pack(F(45), drpk, 128, 48, 128, 48, 128, 1, 0, 48, 128, 0);
  for (int blk = 0; blk < 16; ++blk) {
    pack(F(47) + blk * 2304, rb1pk + blk * 3072, 64, 48, 48, 48, 64, 48, 0, 48, 1, 0);
    pack(F(49) + blk * 2304, rb2pk + blk * 3072, 64, 48, 48, 48, 64, 48, 0, 48, 1, 0);
  }

  Params prm;
  prm.meas = F(0); prm.evnt = F(1); prm.leak = F(2); prm.evlk = F(3);
  prm.pm_w = F(5); prm.pm_b = F(6); prm.pe_w = F(7); prm.pe_b = F(8);
  prm.pl_w = F(9); prm.pl_b = F(10); prm.pel_w = F(11); prm.pel_b = F(12);
  prm.stab_emb = F(13); prm.cyc_emb = F(14);
  prm.er_ln_s = F(15); prm.er_ln_b = F(16); prm.er_fc1_b = F(18); prm.er_fc2_b = F(20);
  prm.ln1_s = F(21); prm.ln1_b = F(22);
  prm.bq = F(25); prm.bk = F(27); prm.bv = F(29); prm.bo = F(31);
  prm.ln2_s = F(32); prm.ln2_b = F(33);
  prm.f1_b = F(35); prm.f2_b = F(37); prm.conv_b = F(39);
  prm.lnf_s = F(40); prm.lnf_b = F(41); prm.Pvec = F(42);
  prm.sc_b = F(44); prm.dr_b = F(46);
  prm.rb1_b = F(48); prm.rb2_b = F(50);
  prm.out_w = F(51); prm.out_b = F(52);
  prm.stab_ids = I(53); prm.cycle_ids = I(54);
  prm.Bp = Bp;
  prm.er1pk = er1pk; prm.er2pk = er2pk;
  prm.wqpk = wqpk; prm.wkpk = wkpk; prm.wvpk = wvpk; prm.wopk = wopk;
  prm.f1pk = f1pk; prm.f2pk = f2pk; prm.convpk = convpk;
  prm.scpk = scpk; prm.drpk = drpk; prm.rb1pk = rb1pk; prm.rb2pk = rb2pk;
  prm.out = (float*)d_out;

  fused_forward<<<dim3(16), dim3(256), 327680, stream>>>(prm);
}
